// DenseRelationDistill_55301998903745
// MI455X (gfx1250) — matrix operand order repaired and checked by reading, not compiled, not run on hardware
//
#include <hip/hip_runtime.h>

// DenseRelationDistill for MI455X (gfx1250, wave32, WMMA).
// All heavy math routed through v_wmma_f32_16x16x32_f16 (f32 accumulate):
//   conv3x3 -> im2col(f16) + GEMM, QK / PV einsums -> GEMM,
//   conv1x1 folded with bilinear upsample + bn:  out = W1@f + bilerp(W2@bn(fin)) + b.
// GEMM: 128-thread block (4 waves), 64x64 tile (MT templated: 2 or 4 M-subtiles,
// uniform per launch so the hot loop has no guards / no acc phi-copies).
// B K-chunk staged through LDS with coalesced 16B loads, zero-filled edges;
// A fragments loaded with global_load_b128 (in-loop index arithmetic keeps the
// global address space -- carried pointers regressed to flat_load in round 3).

#define BS     2
#define NCLS   15
#define INDIM  256
#define KEYDIM 32
#define VALDIM 128
#define HW     1024   // 32*32 attention grid
#define K9     2304   // 256*9 im2col depth

#define TN 64         // block tile N
#define BSTRIDE 72    // LDS row stride (halfs): 144B, keeps 16B store alignment

typedef __attribute__((ext_vector_type(16))) _Float16 v16h;
typedef __attribute__((ext_vector_type(8)))  _Float16 v8h;
typedef __attribute__((ext_vector_type(8)))  float    v8f;

// ---------------- small utility kernels ----------------

__global__ void cvt_f16_kernel(const float* __restrict__ in, _Float16* __restrict__ out,
                               long long n)
{
    long long i = (long long)blockIdx.x * blockDim.x + threadIdx.x;
    if (i < n) out[i] = (_Float16)in[i];
}

__global__ void split_combine_kernel(const float* __restrict__ cw,
                                     _Float16* __restrict__ w1, _Float16* __restrict__ w2)
{
    int i = blockIdx.x * blockDim.x + threadIdx.x;
    if (i >= INDIM * 2 * INDIM) return;
    int o = i >> 9, c = i & 511;
    _Float16 v = (_Float16)cw[i];
    if (c < INDIM) w1[o * INDIM + c] = v;
    else           w2[o * INDIM + (c - INDIM)] = v;
}

// in: (Z,256,32,32) f32  ->  out: (Z,2304,1024) f16   (zero-padded 3x3 im2col)
__global__ void im2col3x3_kernel(const float* __restrict__ in, _Float16* __restrict__ out, int Z)
{
    long long i = (long long)blockIdx.x * blockDim.x + threadIdx.x;
    long long total = (long long)Z * K9 * HW;
    if (i >= total) return;
    int s = (int)(i & (HW - 1));
    long long r = i >> 10;
    int kidx = (int)(r % K9);
    int z = (int)(r / K9);
    int ci = kidx / 9, k9 = kidx % 9;
    int ky = k9 / 3, kx = k9 % 3;
    int y = s >> 5, x = s & 31;
    int iy = y + ky - 1, ix = x + kx - 1;
    float v = 0.f;
    if (iy >= 0 && iy < 32 && ix >= 0 && ix < 32)
        v = in[(((long long)z * INDIM + ci) * 32 + iy) * 32 + ix];
    out[i] = (_Float16)v;
}

// align-corners bilinear downsample (BS,256,H,W) -> (BS,256,32,32)
__global__ void resize_to32_kernel(const float* __restrict__ in, float* __restrict__ out,
                                   int H, int W)
{
    int i = blockIdx.x * blockDim.x + threadIdx.x;
    if (i >= BS * INDIM * HW) return;
    int s = i & (HW - 1);
    int bc = i >> 10;
    int oy = s >> 5, ox = s & 31;
    float sy = (float)oy * (float)(H - 1) / 31.0f;
    float sx = (float)ox * (float)(W - 1) / 31.0f;
    int y0 = (int)sy, x0 = (int)sx;
    int y1 = y0 + 1 < H ? y0 + 1 : H - 1;
    int x1 = x0 + 1 < W ? x0 + 1 : W - 1;
    float wy = sy - (float)y0, wx = sx - (float)x0;
    const float* p = in + (long long)bc * H * W;
    float tl = p[y0 * W + x0], tr = p[y0 * W + x1];
    float bl = p[y1 * W + x0], br = p[y1 * W + x1];
    out[i] = (tl * (1.f - wx) + tr * wx) * (1.f - wy) + (bl * (1.f - wx) + br * wx) * wy;
}

// per z: in f32 (R,C) -> out f16 (C,R)
__global__ void transpose_cvt_kernel(const float* __restrict__ in, _Float16* __restrict__ out,
                                     int R, int C, long long total)
{
    long long i = (long long)blockIdx.x * blockDim.x + threadIdx.x;
    if (i >= total) return;
    int c = (int)(i % C);
    long long t = i / C;
    int r = (int)(t % R);
    int z = (int)(t / R);
    out[((long long)z * C + c) * R + r] = (_Float16)in[((long long)z * R + r) * C + c];
}

// PT: (30 pairs, 1024 s, 1024 q) f32. One block per (pair,s) row; softmax over q,
// store transposed as f16: Pn[pair][q][s]  (ready as B-operand of the PV GEMM).
__global__ void softmax_T_kernel(const float* __restrict__ PT, _Float16* __restrict__ Pn)
{
    int rowId = blockIdx.x;
    int pair = rowId >> 10;
    int s = rowId & 1023;
    const float* row = PT + (long long)rowId * 1024;
    __shared__ float red[256];
    int tid = threadIdx.x;
    float vals[4];
    float mx = -3.4e38f;
#pragma unroll
    for (int i = 0; i < 4; ++i) {
        vals[i] = row[tid + i * 256];
        mx = fmaxf(mx, vals[i]);
    }
    red[tid] = mx; __syncthreads();
    for (int o = 128; o > 0; o >>= 1) {
        if (tid < o) red[tid] = fmaxf(red[tid], red[tid + o]);
        __syncthreads();
    }
    mx = red[0]; __syncthreads();
    float sum = 0.f;
#pragma unroll
    for (int i = 0; i < 4; ++i) { vals[i] = __expf(vals[i] - mx); sum += vals[i]; }
    red[tid] = sum; __syncthreads();
    for (int o = 128; o > 0; o >>= 1) {
        if (tid < o) red[tid] += red[tid + o];
        __syncthreads();
    }
    float inv = 1.f / red[0];
#pragma unroll
    for (int i = 0; i < 4; ++i)
        Pn[(long long)pair * (HW * HW) + (long long)(tid + i * 256) * HW + s] =
            (_Float16)(vals[i] * inv);
}

// fin = bn(concat([NCLS*vq, agg]))  -> f16 (BS,256,1024)
__global__ void fin_build_kernel(const float* __restrict__ vq, const float* __restrict__ agg,
                                 const float* __restrict__ gamma, const float* __restrict__ beta,
                                 const float* __restrict__ mean, const float* __restrict__ var,
                                 int idx, _Float16* __restrict__ fin)
{
    int i = blockIdx.x * blockDim.x + threadIdx.x;
    if (i >= BS * INDIM * HW) return;
    int s = i & 1023;
    int c = (i >> 10) & 255;
    int b = i >> 18;
    float v;
    if (c < VALDIM) v = (float)NCLS * vq[((b * VALDIM + c) << 10) + s];
    else            v = agg[((b * VALDIM + (c - VALDIM)) << 10) + s];
    float g  = gamma[idx * INDIM + c], be = beta[idx * INDIM + c];
    float mu = mean[idx * INDIM + c],  va = var[idx * INDIM + c];
    v = g * (v - mu) * rsqrtf(va + 1e-5f) + be;
    fin[i] = (_Float16)v;
}

// ---------------- generic WMMA GEMM ----------------
// C[z][M,N] (f32, row-major) = sum_{t<T} A16[aoff(z,t)][M,K] @ B16[boff(z,t)][K,N] (+bias[m])
// optional epilogue: += align-corners bilinear upsample of g[z][m] (32x32) at pixel n.
// Block: 128 threads = 4 waves; tile (MT*16)M x 64N. MT is a template constant
// (uniform per launch: every call site has M == 32 or M % 64 == 0), so the
// mi loop unrolls with no guards and accumulators never get phi-copied.
// B K-chunk (32x64) staged through LDS, edges zero-filled at staging so all
// fragment reads and WMMAs run with EXEC all-1s.
template <int MT>
__global__ __launch_bounds__(128)
void gemm_wmma_kernel(const _Float16* __restrict__ A, const _Float16* __restrict__ B,
                      float* __restrict__ C,
                      int N, int K, int lda, int ldb, int ldc,
                      int T,
                      long long aZstr, int aZmod, long long aTstr,
                      long long bZstr, int bZdiv, long long bTstr,
                      long long cZstr,
                      const float* __restrict__ bias,
                      const float* __restrict__ g, long long gZstr,
                      int outH, int outW)
{
    __shared__ _Float16 Bs[32][BSTRIDE];   // K-chunk x N-tile (row-major, padded)

    int z = blockIdx.z;
    int m0 = blockIdx.y * (MT * 16);
    int n0 = blockIdx.x * TN;
    int tid = threadIdx.x;
    int wave = tid >> 5;
    int lane = tid & 31;
    int half = lane >> 4;
    int r16 = lane & 15;

    int nLocal = wave * 16 + r16;          // 0..63 column within tile
    int n = n0 + nLocal;

    const _Float16* Az = A + (long long)(z % aZmod) * aZstr;
    const _Float16* Bz = B + (long long)(z / bZdiv) * bZstr;

    // staging coords: thread -> (k row, 16-half segment)
    int sk = tid >> 2;                     // 0..31
    int sn = (tid & 3) * 16;               // 0,16,32,48

    v8f zero8 = {0.f, 0.f, 0.f, 0.f, 0.f, 0.f, 0.f, 0.f};
    v8f acc[MT];
#pragma unroll
    for (int i = 0; i < MT; ++i) acc[i] = zero8;

    bool fastB = ((ldb & 7) == 0) && (n0 + TN <= N);   // 16B-aligned full-width path

    for (int t = 0; t < T; ++t) {
        // A base for this t: lane-invariant part + per-lane row offset folded in
        const _Float16* At = Az + (long long)t * aTstr + (long long)(m0 + r16) * lda + half * 8;
        const _Float16* Bt = Bz + (long long)t * bTstr;
        for (int kb = 0; kb < K; kb += 32) {
            // ---- stage B chunk (32 x 64) into LDS, coalesced ----
            __syncthreads();
            const _Float16* src = Bt + (long long)(kb + sk) * ldb + (n0 + sn);
            if (fastB) {
                *(v8h*)&Bs[sk][sn]     = *(const v8h*)(src);
                *(v8h*)&Bs[sk][sn + 8] = *(const v8h*)(src + 8);
            } else {
#pragma unroll
                for (int e = 0; e < 16; ++e) {
                    int nn = n0 + sn + e;
                    Bs[sk][sn + e] = (nn < N) ? src[e] : (_Float16)0.f;
                }
            }
            if (kb + 32 < K)
                __builtin_prefetch(src + (long long)32 * ldb, 0, 1);  // global_prefetch_b8
            __syncthreads();
            // ---- B fragment from LDS: element j <-> K = half*8 + j (j<8), 16 + half*8 + (j-8) (j>=8) ----
            v16h bfrag;
#pragma unroll
            for (int j = 0; j < 16; ++j)
                bfrag[j] = Bs[half * 8 + j + (j >= 8 ? 8 : 0)][nLocal];
            // ---- MT M-subtiles share this B fragment ----
#pragma unroll
            for (int mi = 0; mi < MT; ++mi) {
                // A fragment: lane(half,m): K chunks [half*8..+7], [half*8+16..+7]
                // (address computed per iteration -> stays a global_load_b128)
                const _Float16* ap = At + (long long)(mi * 16) * lda + kb;
                union { v16h v; v8h h[2]; } ua;
                ua.h[0] = *(const v8h*)(ap);
                ua.h[1] = *(const v8h*)(ap + 16);
                acc[mi] = __builtin_amdgcn_wmma_f32_16x16x32_f16(
                    false, ua.v, false, bfrag, (short)0, acc[mi], false, false);
            }
        }
    }
    if (n >= N) return;
#pragma unroll
    for (int mi = 0; mi < MT; ++mi) {
#pragma unroll
        for (int r = 0; r < 8; ++r) {
            int m = m0 + mi * 16 + half * 8 + r;
            float v = acc[mi][r];
            if (bias) v += bias[m];
            if (g) {
                int y = n / outW, x = n % outW;
                float sy = (outH > 1) ? (float)y * 31.0f / (float)(outH - 1) : 0.f;
                float sx = (outW > 1) ? (float)x * 31.0f / (float)(outW - 1) : 0.f;
                int y0 = (int)sy, x0 = (int)sx;
                int y1 = y0 + 1 < 32 ? y0 + 1 : 31;
                int x1 = x0 + 1 < 32 ? x0 + 1 : 31;
                float wy = sy - (float)y0, wx = sx - (float)x0;
                const float* gp = g + (long long)z * gZstr + ((long long)m << 10);
                float tl = gp[(y0 << 5) + x0], tr = gp[(y0 << 5) + x1];
                float bl = gp[(y1 << 5) + x0], br = gp[(y1 << 5) + x1];
                v += (tl * (1.f - wx) + tr * wx) * (1.f - wy) +
                     (bl * (1.f - wx) + br * wx) * wy;
            }
            C[(long long)z * cZstr + (long long)m * ldc + n] = v;
        }
    }
}

static void gemm_launch(hipStream_t st, const _Float16* A, const _Float16* B, float* C,
                        int M, int N, int K, int lda, int ldb, int ldc, int Z, int T,
                        long long aZstr, int aZmod, long long aTstr,
                        long long bZstr, int bZdiv, long long bTstr, long long cZstr,
                        const float* bias, const float* g, long long gZstr,
                        int outH, int outW)
{
    // Every call site has M == 32 (-> MT=2) or M % 64 == 0 (-> MT=4), so MT is
    // uniform across the grid and can be a template constant.
    if (M >= 64) {
        dim3 grid((N + TN - 1) / TN, M / 64, Z);
        gemm_wmma_kernel<4><<<grid, dim3(128), 0, st>>>(A, B, C, N, K, lda, ldb, ldc, T,
                                                        aZstr, aZmod, aTstr, bZstr, bZdiv,
                                                        bTstr, cZstr, bias, g, gZstr,
                                                        outH, outW);
    } else {
        dim3 grid((N + TN - 1) / TN, 1, Z);
        gemm_wmma_kernel<2><<<grid, dim3(128), 0, st>>>(A, B, C, N, K, lda, ldb, ldc, T,
                                                        aZstr, aZmod, aTstr, bZstr, bZdiv,
                                                        bTstr, cZstr, bias, g, gZstr,
                                                        outH, outW);
    }
}

// ---------------- launcher ----------------

extern "C" void kernel_launch(void* const* d_in, const int* in_sizes, int n_in,
                              void* d_out, int out_size, void* d_ws, size_t ws_size,
                              hipStream_t stream)
{
    static const int LH[5] = {200, 100, 50, 25, 13};
    static const int LW[5] = {336, 168, 84, 42, 21};

    const float* f[5];
    for (int i = 0; i < 5; ++i) f[i] = (const float*)d_in[i];
    const float* attn      = (const float*)d_in[5];
    const float* key_t_w   = (const float*)d_in[6];
    const float* key_t_b   = (const float*)d_in[7];
    const float* value_t_w = (const float*)d_in[8];
    const float* value_t_b = (const float*)d_in[9];
    const float* key_q_w   = (const float*)d_in[10];
    const float* key_q_b   = (const float*)d_in[11];
    const float* value_q_w = (const float*)d_in[12];
    const float* value_q_b = (const float*)d_in[13];
    const float* bn_gamma  = (const float*)d_in[14];
    const float* bn_beta   = (const float*)d_in[15];
    const float* bn_mean   = (const float*)d_in[16];
    const float* bn_var    = (const float*)d_in[17];
    const float* combine_w = (const float*)d_in[18];
    const float* combine_b = (const float*)d_in[19];
    float* out = (float*)d_out;
    (void)in_sizes; (void)n_in; (void)out_size; (void)ws_size;

    size_t off = 0;
    auto alloc = [&](size_t bytes) -> char* {
        off = (off + 255) & ~(size_t)255;
        char* p = (char*)d_ws + off;
        off += bytes;
        return p;
    };

    _Float16* colA16 = (_Float16*)alloc((size_t)NCLS * K9 * HW * 2);
    _Float16* Wk16   = (_Float16*)alloc((size_t)KEYDIM * K9 * 2);
    _Float16* Wv16   = (_Float16*)alloc((size_t)VALDIM * K9 * 2);
    _Float16* Wkq16  = (_Float16*)alloc((size_t)5 * KEYDIM * K9 * 2);
    _Float16* Wvq16  = (_Float16*)alloc((size_t)5 * VALDIM * K9 * 2);
    _Float16* W1_16  = (_Float16*)alloc((size_t)INDIM * INDIM * 2);
    _Float16* W2_16  = (_Float16*)alloc((size_t)INDIM * INDIM * 2);
    float*    kt_f   = (float*)alloc((size_t)NCLS * KEYDIM * HW * 4);
    _Float16* ktT16  = (_Float16*)alloc((size_t)NCLS * KEYDIM * HW * 2);
    float*    vt_f   = (float*)alloc((size_t)NCLS * VALDIM * HW * 4);
    _Float16* vt16   = (_Float16*)alloc((size_t)NCLS * VALDIM * HW * 2);
    float*    fr_f   = (float*)alloc((size_t)BS * INDIM * HW * 4);
    _Float16* colF16 = (_Float16*)alloc((size_t)BS * K9 * HW * 2);
    float*    kq_f   = (float*)alloc((size_t)BS * KEYDIM * HW * 4);
    _Float16* kq16   = (_Float16*)alloc((size_t)BS * KEYDIM * HW * 2);
    float*    vq_f   = (float*)alloc((size_t)BS * VALDIM * HW * 4);
    float*    agg_f  = (float*)alloc((size_t)BS * VALDIM * HW * 4);
    _Float16* fin16  = (_Float16*)alloc((size_t)BS * INDIM * HW * 2);
    float*    g_f    = (float*)alloc((size_t)BS * INDIM * HW * 4);
    _Float16* Pn16   = (_Float16*)alloc((size_t)BS * NCLS * HW * HW * 2);
    // PT (f32 logits) and Ff16 (f16 copy of level features) are never live
    // simultaneously within a level -> share one region.
    size_t ptBytes = (size_t)BS * NCLS * HW * HW * 4;
    size_t ffBytes = (size_t)BS * INDIM * 200 * 336 * 2;
    char* region = alloc(ptBytes > ffBytes ? ptBytes : ffBytes);
    float*    PT_f = (float*)region;
    _Float16* Ff16 = (_Float16*)region;

    auto cvt = [&](const float* src, _Float16* dst, long long n) {
        cvt_f16_kernel<<<(int)((n + 255) / 256), 256, 0, stream>>>(src, dst, n);
    };

    // ---- one-time: weight conversion, template keys/values ----
    cvt(key_t_w,   Wk16,  (long long)KEYDIM * K9);
    cvt(value_t_w, Wv16,  (long long)VALDIM * K9);
    cvt(key_q_w,   Wkq16, (long long)5 * KEYDIM * K9);
    cvt(value_q_w, Wvq16, (long long)5 * VALDIM * K9);
    split_combine_kernel<<<(INDIM * 2 * INDIM + 255) / 256, 256, 0, stream>>>(combine_w, W1_16, W2_16);

    {
        long long tot = (long long)NCLS * K9 * HW;
        im2col3x3_kernel<<<(int)((tot + 255) / 256), 256, 0, stream>>>(attn, colA16, NCLS);
    }
    // kt[n](32,1024) = Wk(32,2304) @ colA[n]  (+key_t_b)
    gemm_launch(stream, Wk16, colA16, kt_f, KEYDIM, HW, K9, K9, HW, HW,
                NCLS, 1, 0, 1, 0, (long long)K9 * HW, 1, 0, (long long)KEYDIM * HW,
                key_t_b, nullptr, 0, 0, 0);
    // vt[n](128,1024) = Wv @ colA[n]  (+value_t_b)
    gemm_launch(stream, Wv16, colA16, vt_f, VALDIM, HW, K9, K9, HW, HW,
                NCLS, 1, 0, 1, 0, (long long)K9 * HW, 1, 0, (long long)VALDIM * HW,
                value_t_b, nullptr, 0, 0, 0);
    {
        long long tot = (long long)NCLS * KEYDIM * HW;
        transpose_cvt_kernel<<<(int)((tot + 255) / 256), 256, 0, stream>>>(kt_f, ktT16, KEYDIM, HW, tot);
    }
    cvt(vt_f, vt16, (long long)NCLS * VALDIM * HW);

    long long outOff = 0;
    for (int idx = 0; idx < 5; ++idx) {
        int H = LH[idx], W = LW[idx], P = H * W;
        resize_to32_kernel<<<(BS * INDIM * HW + 255) / 256, 256, 0, stream>>>(f[idx], fr_f, H, W);
        {
            long long tot = (long long)BS * K9 * HW;
            im2col3x3_kernel<<<(int)((tot + 255) / 256), 256, 0, stream>>>(fr_f, colF16, BS);
        }
        // kq[b](32,1024), vq[b](128,1024)
        gemm_launch(stream, Wkq16 + (size_t)idx * KEYDIM * K9, colF16, kq_f,
                    KEYDIM, HW, K9, K9, HW, HW,
                    BS, 1, 0, 1, 0, (long long)K9 * HW, 1, 0, (long long)KEYDIM * HW,
                    key_q_b + idx * KEYDIM, nullptr, 0, 0, 0);
        gemm_launch(stream, Wvq16 + (size_t)idx * VALDIM * K9, colF16, vq_f,
                    VALDIM, HW, K9, K9, HW, HW,
                    BS, 1, 0, 1, 0, (long long)K9 * HW, 1, 0, (long long)VALDIM * HW,
                    value_q_b + idx * VALDIM, nullptr, 0, 0, 0);
        cvt(kq_f, kq16, (long long)BS * KEYDIM * HW);
        // PT[z=b*15+n](s,q) = ktT[n](1024,32) @ kq[b](32,1024)
        gemm_launch(stream, ktT16, kq16, PT_f, HW, HW, KEYDIM, KEYDIM, HW, HW,
                    BS * NCLS, 1,
                    (long long)HW * KEYDIM, NCLS, 0,
                    (long long)KEYDIM * HW, NCLS, 0,
                    (long long)HW * HW,
                    nullptr, nullptr, 0, 0, 0);
        // softmax over q per (pair,s); store transposed f16 (q,s)
        softmax_T_kernel<<<BS * NCLS * HW, 256, 0, stream>>>(PT_f, Pn16);
        // agg[b](128,1024) = sum_n vt[n] @ Pn[b*15+n]   (n-sum fused in T-loop)
        gemm_launch(stream, vt16, Pn16, agg_f, VALDIM, HW, HW, HW, HW, HW,
                    BS, NCLS,
                    0, 1, (long long)VALDIM * HW,
                    (long long)NCLS * HW * HW, 1, (long long)HW * HW,
                    (long long)VALDIM * HW,
                    nullptr, nullptr, 0, 0, 0);
        fin_build_kernel<<<(BS * INDIM * HW + 255) / 256, 256, 0, stream>>>(
            vq_f, agg_f, bn_gamma, bn_beta, bn_mean, bn_var, idx, fin16);
        // g[b](256,1024) = W2 @ fin[b]
        gemm_launch(stream, W2_16, fin16, g_f, INDIM, HW, INDIM, INDIM, HW, HW,
                    BS, 1, 0, 1, 0, (long long)INDIM * HW, 1, 0, (long long)INDIM * HW,
                    nullptr, nullptr, 0, 0, 0);
        // out[b](256,P) = W1 @ f16(f[b]) + bilerp(g[b]) + combine_b
        cvt(f[idx], Ff16, (long long)BS * INDIM * P);
        gemm_launch(stream, W1_16, Ff16, out + outOff, INDIM, P, INDIM, INDIM, P, P,
                    BS, 1, 0, 1, 0, (long long)INDIM * P, 1, 0, (long long)INDIM * P,
                    combine_b, g_f, (long long)INDIM * HW, H, W);
        outOff += (long long)BS * INDIM * P;
    }
}
